// AttnBlock_25357486915719
// MI455X (gfx1250) — matrix operand order repaired and checked by reading, not compiled, not run on hardware
//
#include <hip/hip_runtime.h>

typedef __attribute__((ext_vector_type(16))) _Float16 v16h;
typedef __attribute__((ext_vector_type(8)))  float    v8f;

#define B_    2
#define C_    512
#define HW_   4096
#define G_    32
#define CPG_  16
#define GROUP_ELEMS (CPG_*HW_)   /* 65536, contiguous per (b,group) */

__device__ __forceinline__ unsigned lane_id() { return threadIdx.x & 31u; }

__device__ __forceinline__ v8f vzero() {
    v8f z = {0.f,0.f,0.f,0.f,0.f,0.f,0.f,0.f};
    return z;
}

// ---- WMMA fragment loaders (CDNA5 16x16x32 f16 layouts, ISA 7.12.2) ----

// A 16x32 (MxK) from row-major f16, ld in halves.
// lanes 0-15: M=lane, K={0..7,16..23}; lanes 16-31: M=lane-16, K={8..15,24..31}
__device__ __forceinline__ v16h load_a_f16(const _Float16* p, int ld) {
    unsigned l = lane_id();
    const _Float16* r = p + (size_t)(l & 15u) * ld + ((l >> 4) * 8u);
    v16h a;
#pragma unroll
    for (int j = 0; j < 8; ++j)
#pragma unroll
        for (int e = 0; e < 2; ++e)
            a[2*j + e] = r[(j & 3)*2 + e + ((j >> 2)*16)];
    return a;
}

// A 16x32 from row-major f32 (LDS), converting to f16 on load.
__device__ __forceinline__ v16h load_a_f32(const float* p, int ld) {
    unsigned l = lane_id();
    const float* r = p + (size_t)(l & 15u) * ld + ((l >> 4) * 8u);
    v16h a;
#pragma unroll
    for (int j = 0; j < 8; ++j)
#pragma unroll
        for (int e = 0; e < 2; ++e)
            a[2*j + e] = (_Float16)r[(j & 3)*2 + e + ((j >> 2)*16)];
    return a;
}

// B 32x16 (KxN) where B = src^T, src is row-major [N][K] (p -> src[n0][k0]).
// lanes 0-15: N=lane, K={0..7,16..23}; lanes 16-31: N=lane-16, K={8..15,24..31} (matches A).
__device__ __forceinline__ v16h load_b_nt(const _Float16* p, size_t ld) {
    unsigned l = lane_id();
    const _Float16* r = p + (size_t)(l & 15u) * ld + ((l >> 4) * 8u);
    v16h b;
#pragma unroll
    for (int i = 0; i < 16; ++i) b[i] = r[(i & 7) + (i >> 3) * 16];
    return b;
}

__device__ __forceinline__ v8f wmma16(v16h a, v16h b, v8f c) {
    return __builtin_amdgcn_wmma_f32_16x16x32_f16(false, a, false, b,
                                                  (short)0, c, false, false);
}

// ---------------- kernel 0: cast weights to f16 ----------------
__global__ void k_cast_w(const float* wq, const float* wk, const float* wv,
                         const float* wp, _Float16* w16) {
    int i = blockIdx.x * blockDim.x + threadIdx.x;      // 4 * 512*512
    int m = i >> 18;
    int r = i & 0x3FFFF;
    const float* s = (m == 0) ? wq : (m == 1) ? wk : (m == 2) ? wv : wp;
    w16[i] = (_Float16)s[r];
}

// ---------------- kernel 1: group-norm statistics ----------------
__global__ void k_gn_stats(const float* __restrict__ x, float* __restrict__ stats) {
    __shared__ float rs[256], rq[256];
    const size_t base = (size_t)blockIdx.x * GROUP_ELEMS;   // contiguous slab
    const float4* xv = (const float4*)(x + base);
    float s = 0.f, q = 0.f;
    for (int i = threadIdx.x; i < GROUP_ELEMS / 4; i += 256) {
        float4 v = xv[i];
        s += v.x + v.y + v.z + v.w;
        q += v.x*v.x + v.y*v.y + v.z*v.z + v.w*v.w;
    }
    rs[threadIdx.x] = s; rq[threadIdx.x] = q;
    __syncthreads();
    for (int o = 128; o > 0; o >>= 1) {
        if ((int)threadIdx.x < o) {
            rs[threadIdx.x] += rs[threadIdx.x + o];
            rq[threadIdx.x] += rq[threadIdx.x + o];
        }
        __syncthreads();
    }
    if (threadIdx.x == 0) {
        float mean = rs[0] * (1.f / GROUP_ELEMS);
        float var  = rq[0] * (1.f / GROUP_ELEMS) - mean * mean;
        stats[2*blockIdx.x]     = mean;
        stats[2*blockIdx.x + 1] = rsqrtf(var + 1e-6f);
    }
}

// ------- kernel 2: normalize + transpose (b,c,hw) -> tokens (b,hw,c) f16 -------
__global__ void k_norm_t(const float* __restrict__ x, const float* __restrict__ gw,
                         const float* __restrict__ gb, const float* __restrict__ stats,
                         _Float16* __restrict__ t) {
    __shared__ float tile[32][33];
    int b  = blockIdx.z;
    int c0 = blockIdx.y * 32;
    int p0 = blockIdx.x * 32;
    int tx = threadIdx.x, ty = threadIdx.y;
    for (int r = ty; r < 32; r += 8) {
        int c = c0 + r;
        int gi = 2 * (b * G_ + (c >> 4));
        float mean = stats[gi], rstd = stats[gi + 1];
        float v = x[((size_t)(b * C_ + c)) * HW_ + p0 + tx];
        tile[r][tx] = (v - mean) * rstd * gw[c] + gb[c];
    }
    __syncthreads();
    for (int r = ty; r < 32; r += 8) {
        t[((size_t)(b * HW_ + p0 + r)) * C_ + c0 + tx] = (_Float16)tile[tx][r];
    }
}

// ------- kernel 3: QKV projections, q = scale*(t @ Wq^T + bq), etc. -------
// grid (M/128=64, N/128=4, 3); block 256 = 8 waves; wave tile 32x64.
// z==2 (V) is stored TRANSPOSED as vT[b][c][hw] so the attention PV GEMM's
// B fragments are per-lane loads of two contiguous 8-half runs.
__global__ void __launch_bounds__(256)
k_qkv(const _Float16* __restrict__ t, const _Float16* __restrict__ w16,
      const float* __restrict__ bq, const float* __restrict__ bk,
      const float* __restrict__ bv,
      _Float16* __restrict__ q, _Float16* __restrict__ k, _Float16* __restrict__ vt) {
    int z = blockIdx.z;
    const _Float16* w = w16 + (size_t)z * C_ * C_;
    const float* bias = (z == 0) ? bq : (z == 1) ? bk : bv;
    const float scale = (z == 0) ? 0.044194173824159216f : 1.0f; // 512^-0.5 folded into Q

    unsigned wv_ = threadIdx.x >> 5;
    unsigned l   = lane_id();
    int wm = wv_ >> 1, wn = wv_ & 1;
    int m0 = blockIdx.x * 128 + wm * 32;
    int n0 = blockIdx.y * 128 + wn * 64;

    v8f acc[2][4];
#pragma unroll
    for (int i = 0; i < 2; ++i)
#pragma unroll
        for (int j = 0; j < 4; ++j) acc[i][j] = vzero();

    for (int c0 = 0; c0 < C_; c0 += 32) {
        v16h a0 = load_a_f16(t + (size_t)m0 * C_ + c0, C_);
        v16h a1 = load_a_f16(t + (size_t)(m0 + 16) * C_ + c0, C_);
#pragma unroll
        for (int j = 0; j < 4; ++j) {
            v16h b = load_b_nt(w + (size_t)(n0 + 16*j) * C_ + c0, C_);
            acc[0][j] = wmma16(a0, b, acc[0][j]);
            acc[1][j] = wmma16(a1, b, acc[1][j]);
        }
    }
    int nn = l & 15, mh = (l >> 4) * 8;
    if (z == 2) {
        // V: scatter transposed -> vT[(b*C + col)*HW + hw], 8 contiguous halves/lane
#pragma unroll
        for (int i = 0; i < 2; ++i)
#pragma unroll
            for (int j = 0; j < 4; ++j) {
                int col = n0 + 16*j + nn;
                float bc = bias[col];
#pragma unroll
                for (int r = 0; r < 8; ++r) {
                    int row = m0 + 16*i + mh + r;     // global token index
                    int b2  = row >> 12;
                    int p   = row & 4095;
                    vt[((size_t)(b2 * C_ + col)) * HW_ + p] =
                        (_Float16)(acc[i][j][r] + bc);
                }
            }
    } else {
        _Float16* out = (z == 0) ? q : k;
#pragma unroll
        for (int i = 0; i < 2; ++i)
#pragma unroll
            for (int j = 0; j < 4; ++j) {
                int col = n0 + 16*j + nn;
                float bc = bias[col];
#pragma unroll
                for (int r = 0; r < 8; ++r) {
                    int row = m0 + 16*i + mh + r;
                    out[(size_t)row * C_ + col] = (_Float16)((acc[i][j][r] + bc) * scale);
                }
            }
    }
}

// ------- kernel 4: attention, 16 query rows per block, full score row in LDS -------
// LDS: S[16][4096] f32 (256KB) + Qt[16][512] f16 (16KB) + reductions (~1.1KB) < 320KB/WGP
__global__ void __launch_bounds__(256)
k_attn(const _Float16* __restrict__ q, const _Float16* __restrict__ k,
       const _Float16* __restrict__ vt, _Float16* __restrict__ o) {
    extern __shared__ char smem[];
    float*    S   = (float*)smem;                                   // 16*4096 f32
    _Float16* Qt  = (_Float16*)(smem + (size_t)16 * HW_ * 4);       // 16*512 f16
    float*    red = (float*)(smem + (size_t)16 * HW_ * 4 + 16 * C_ * 2); // 256
    float*    rst = red + 256;                                      // 16

    int m0 = blockIdx.x * 16;
    int bb = blockIdx.y;
    const _Float16* qb  = q  + (size_t)bb * HW_ * C_;
    const _Float16* kb  = k  + (size_t)bb * HW_ * C_;
    const _Float16* vtb = vt + (size_t)bb * C_ * HW_;   // [C][HW]
    int tid = threadIdx.x;
    unsigned wv_ = tid >> 5;

    // load Q tile (16x512 halves, contiguous 16KB)
    {
        const int4* src = (const int4*)(qb + (size_t)m0 * C_);
        int4* dst = (int4*)Qt;
        for (int i = tid; i < 16 * C_ / 8; i += 256) dst[i] = src[i];
    }
    __syncthreads();

    // ---- phase A: S = Q @ K^T (scale already folded into Q) ----
    for (int jc = 0; jc < 8; ++jc) {
        int j0 = (int)wv_ * 512 + jc * 64;
        if (jc < 7) {
            // warm the next chunk's K rows (global_prefetch_b8)
            unsigned l = lane_id();
            __builtin_prefetch(kb + (size_t)(j0 + 64 + (int)l * 2) * C_, 0, 1);
            __builtin_prefetch(kb + (size_t)(j0 + 64 + (int)l * 2 + 1) * C_, 0, 1);
        }
        v8f acc[4];
#pragma unroll
        for (int u = 0; u < 4; ++u) acc[u] = vzero();
        for (int c0 = 0; c0 < C_; c0 += 32) {
            v16h a = load_a_f16(Qt + c0, C_);
#pragma unroll
            for (int u = 0; u < 4; ++u) {
                v16h b = load_b_nt(kb + (size_t)(j0 + 16*u) * C_ + c0, C_);
                acc[u] = wmma16(a, b, acc[u]);
            }
        }
        unsigned l = lane_id();
        int nn = l & 15, mh = (l >> 4) * 8;
#pragma unroll
        for (int u = 0; u < 4; ++u)
#pragma unroll
            for (int r = 0; r < 8; ++r)
                S[(size_t)(mh + r) * HW_ + j0 + 16*u + nn] = acc[u][r];
    }
    __syncthreads();

    // ---- phase B: row softmax over S[16][4096], 16 threads per row ----
    int rr = tid >> 4, ss = tid & 15;
    float mx = -3.0e38f;
    for (int j = ss; j < HW_; j += 16) mx = fmaxf(mx, S[(size_t)rr * HW_ + j]);
    red[tid] = mx;
    __syncthreads();
    if (ss == 0) {
        float m2 = red[rr * 16];
        for (int i = 1; i < 16; ++i) m2 = fmaxf(m2, red[rr * 16 + i]);
        rst[rr] = m2;
    }
    __syncthreads();
    float rm = rst[rr];
    float sum = 0.f;
    for (int j = ss; j < HW_; j += 16) {
        float e = __expf(S[(size_t)rr * HW_ + j] - rm);
        S[(size_t)rr * HW_ + j] = e;
        sum += e;
    }
    __syncthreads();
    red[tid] = sum;
    __syncthreads();
    if (ss == 0) {
        float s2 = 0.f;
        for (int i = 0; i < 16; ++i) s2 += red[rr * 16 + i];
        rst[rr] = 1.0f / s2;
    }
    __syncthreads();
    float inv = rst[rr];
    for (int j = ss; j < HW_; j += 16) S[(size_t)rr * HW_ + j] *= inv;
    __syncthreads();

    // ---- phase C: O = P @ V, each wave owns 64 output channels ----
    // V is stored transposed [C][HW], so B fragments are contiguous per lane.
    {
        int c0w = (int)wv_ * 64;
        v8f acc[4];
#pragma unroll
        for (int u = 0; u < 4; ++u) acc[u] = vzero();
        for (int k0 = 0; k0 < HW_; k0 += 32) {
            if (k0 + 256 < HW_) {
                unsigned l = lane_id();
                __builtin_prefetch(vtb + (size_t)(c0w + (int)l) * HW_ + k0 + 256, 0, 1);
                __builtin_prefetch(vtb + (size_t)(c0w + 32 + (int)l) * HW_ + k0 + 256, 0, 1);
            }
            v16h a = load_a_f32(S + k0, HW_);   // P (f32 LDS) -> f16 fragment
#pragma unroll
            for (int u = 0; u < 4; ++u) {
                v16h b = load_b_nt(vtb + (size_t)(c0w + 16*u) * HW_ + k0, HW_);
                acc[u] = wmma16(a, b, acc[u]);
            }
        }
        unsigned l = lane_id();
        int nn = l & 15, mh = (l >> 4) * 8;
#pragma unroll
        for (int u = 0; u < 4; ++u)
#pragma unroll
            for (int r = 0; r < 8; ++r)
                o[(size_t)(bb * HW_ + m0 + mh + r) * C_ + c0w + 16*u + nn] =
                    (_Float16)acc[u][r];
    }
}

// ------- kernel 5: out = x + (o_tok @ Wp^T + bp), transposed back to (b,c,h,w) -------
__global__ void __launch_bounds__(256)
k_proj(const _Float16* __restrict__ ot, const _Float16* __restrict__ wp16,
       const float* __restrict__ bp, const float* __restrict__ x,
       float* __restrict__ out) {
    unsigned wv_ = threadIdx.x >> 5;
    unsigned l   = lane_id();
    int wm = wv_ >> 1, wn = wv_ & 1;
    int m0 = blockIdx.x * 128 + wm * 32;
    int n0 = blockIdx.y * 128 + wn * 64;

    v8f acc[2][4];
#pragma unroll
    for (int i = 0; i < 2; ++i)
#pragma unroll
        for (int j = 0; j < 4; ++j) acc[i][j] = vzero();

    for (int c0 = 0; c0 < C_; c0 += 32) {
        v16h a0 = load_a_f16(ot + (size_t)m0 * C_ + c0, C_);
        v16h a1 = load_a_f16(ot + (size_t)(m0 + 16) * C_ + c0, C_);
#pragma unroll
        for (int j = 0; j < 4; ++j) {
            v16h b = load_b_nt(wp16 + (size_t)(n0 + 16*j) * C_ + c0, C_);
            acc[0][j] = wmma16(a0, b, acc[0][j]);
            acc[1][j] = wmma16(a1, b, acc[1][j]);
        }
    }
    int nn = l & 15, mh = (l >> 4) * 8;
#pragma unroll
    for (int i = 0; i < 2; ++i)
#pragma unroll
        for (int j = 0; j < 4; ++j) {
            int col = n0 + 16*j + nn;
            float bc = bp[col];
#pragma unroll
            for (int r = 0; r < 8; ++r) {
                int row = m0 + 16*i + mh + r;      // global token index
                int b   = row >> 12;               // / 4096
                int p   = row & 4095;
                size_t oi = ((size_t)(b * C_ + col)) * HW_ + p;
                out[oi] = x[oi] + acc[i][j][r] + bc;
            }
        }
}

// ---------------- launcher ----------------
extern "C" void kernel_launch(void* const* d_in, const int* in_sizes, int n_in,
                              void* d_out, int out_size, void* d_ws, size_t ws_size,
                              hipStream_t stream) {
    const float* x  = (const float*)d_in[0];
    const float* gw = (const float*)d_in[1];
    const float* gb = (const float*)d_in[2];
    const float* wq = (const float*)d_in[3];
    const float* bq = (const float*)d_in[4];
    const float* wk = (const float*)d_in[5];
    const float* bk = (const float*)d_in[6];
    const float* wv = (const float*)d_in[7];
    const float* bv = (const float*)d_in[8];
    const float* wp = (const float*)d_in[9];
    const float* bp = (const float*)d_in[10];
    float* out = (float*)d_out;

    const size_t TOK = (size_t)B_ * HW_ * C_;   // 4M elements
    _Float16* t   = (_Float16*)d_ws;
    _Float16* q   = t + TOK;
    _Float16* k   = q + TOK;
    _Float16* vt  = k + TOK;                    // V stored transposed [b][c][hw]
    _Float16* o   = vt + TOK;
    _Float16* w16 = o + TOK;                    // 4 * C*C halves
    float*  stats = (float*)(w16 + (size_t)4 * C_ * C_);

    k_cast_w<<<dim3(4 * C_ * C_ / 256), dim3(256), 0, stream>>>(wq, wk, wv, wp, w16);
    k_gn_stats<<<dim3(B_ * G_), dim3(256), 0, stream>>>(x, stats);
    k_norm_t<<<dim3(HW_ / 32, C_ / 32, B_), dim3(32, 8), 0, stream>>>(x, gw, gb, stats, t);
    k_qkv<<<dim3(B_ * HW_ / 128, C_ / 128, 3), dim3(256), 0, stream>>>(
        t, w16, bq, bk, bv, q, k, vt);

    size_t smem = (size_t)16 * HW_ * sizeof(float)      // S
                + (size_t)16 * C_  * sizeof(_Float16)   // Q tile
                + (256 + 16) * sizeof(float);           // reductions
    k_attn<<<dim3(HW_ / 16, B_), dim3(256), smem, stream>>>(q, k, vt, o);

    k_proj<<<dim3(B_ * HW_ / 128, C_ / 128), dim3(256), 0, stream>>>(
        o, w16 + (size_t)3 * C_ * C_, bp, x, out);
}
